// _ContextKnowledgeEncoder_44805098832164
// MI455X (gfx1250) — matrix operand order repaired and checked by reading, not compiled, not run on hardware
//
#include <hip/hip_runtime.h>
#include <hip/hip_bf16.h>

typedef __attribute__((ext_vector_type(16))) __bf16 v16bf;
typedef __attribute__((ext_vector_type(8)))  __bf16 bf16x8;
typedef __attribute__((ext_vector_type(8)))  float  v8f;
typedef __attribute__((ext_vector_type(4)))  int    v4i;
typedef v4i __attribute__((address_space(1))) gv4i;   // global
typedef v4i __attribute__((address_space(3))) lv4i;   // LDS

#define D_MODEL   1024
#define N_BATCH   8
#define K_KNOW    32
#define TK        128
#define TS        256
#define FULL_T    (TK + TS)            // 384
#define OUT_ENC_ELEMS ((size_t)N_BATCH * FULL_T * D_MODEL)   // 3145728
#define OUT_MASK_ELEMS ((size_t)N_BATCH * FULL_T)            // 3072
#define NEG_INF_F (-1e30f)

// ---------------- ws layout ----------------
#define WS_W16T_OFF   ((size_t)0)
#define WS_CTXSUM_OFF ((size_t)(1024u * 1024u * 2u))
#define WS_KNOWSUM_OFF (WS_CTXSUM_OFF + (size_t)8 * 1024 * 4)
#define WS_IDS_OFF    (WS_KNOWSUM_OFF + (size_t)256 * 1024 * 4)

// CDNA5 async global->LDS DMA path (guarded: falls back to plain ld/st)
#if defined(__gfx1250__) && \
    __has_builtin(__builtin_amdgcn_global_load_async_to_lds_b128) && \
    __has_builtin(__builtin_amdgcn_s_wait_asynccnt)
#define USE_ASYNC_LDS 1
#else
#define USE_ASYNC_LDS 0
#endif

__device__ __forceinline__ void async_b128(const __bf16* g, __bf16* l) {
#if USE_ASYNC_LDS
    __builtin_amdgcn_global_load_async_to_lds_b128((gv4i*)g, (lv4i*)l, 0, 0);
#else
    (void)g; (void)l;
#endif
}

__device__ __forceinline__ void async_wait0() {
#if USE_ASYNC_LDS
    __builtin_amdgcn_s_wait_asynccnt(0);
#endif
}

__device__ __forceinline__ float gelu_tanh(float x) {
    const float c = 0.7978845608028654f;   // sqrt(2/pi)
    float x3 = x * x * x;
    return 0.5f * x * (1.0f + tanhf(c * (x + 0.044715f * x3)));
}

// W16T[n*1024 + k] = (bf16) W_proj[k*1024 + n]
__global__ __launch_bounds__(256) void convert_w_kernel(const float* __restrict__ Wp,
                                                        __bf16* __restrict__ W16T) {
    int idx = blockIdx.x * 256 + threadIdx.x;     // 0 .. 1M-1
    int n = idx >> 10;
    int k = idx & 1023;
    W16T[idx] = (__bf16)Wp[k * 1024 + n];
}

__global__ __launch_bounds__(256) void zero_ctx_kernel(float* __restrict__ p) {
    p[blockIdx.x * 256 + threadIdx.x] = 0.0f;     // 8192 floats
}

// ------------------------------------------------------------------
// Fused embed-gather + GEMM (bf16 WMMA, f32 acc) + gelu + mask.
// Block tile M=128, N=128, K-step 32; double-buffered LDS; B tile via
// async global->LDS DMA; A tile register-prefetched + f32->bf16 convert.
// 8 waves: wi(0..3) x wj(0..1); each wave: 2x4 WMMA tiles of 16x16.
// MODE 0: ctx encode  -> full_enc[:,128:,:] + atomic ctx row-sums
// MODE 1: know encode -> per-chunk pooled sums only (Tk == M-tile)
// MODE 2: selected chunk re-encode -> full_enc[:,:128,:]
// ------------------------------------------------------------------
template <int MODE>
__global__ __launch_bounds__(256) void encode_gemm(const int* __restrict__ tokens,
                                                   const float* __restrict__ W_emb,
                                                   const __bf16* __restrict__ W16T,
                                                   float* __restrict__ out,
                                                   float* __restrict__ ctx_sum,
                                                   float* __restrict__ know_sum,
                                                   const int* __restrict__ ids) {
    __shared__ __bf16 Al[2][128 * 40];   // [m][k], stride 40 halves (80B, 16B aligned)
    __shared__ __bf16 Bl[2][128 * 40];   // [n][k]
    __shared__ int    tokS[128];
    __shared__ float  colsum[128];

    const int tid   = threadIdx.x;
    const int mtile = blockIdx.x;
    const int ntile = blockIdx.y;

    if (tid < 128) {
        int tk;
        if (MODE == 2) {
            int id = ids[mtile];                       // mtile == batch n
            tk = tokens[(mtile * K_KNOW + id) * TK + tid];
        } else {
            tk = tokens[mtile * 128 + tid];            // flat rows
        }
        tokS[tid]   = tk;
        colsum[tid] = 0.0f;
    }
    __syncthreads();

    const int lane   = tid & 31;
    const int wid    = tid >> 5;
    const int wi     = wid & 3;        // M wave coord (0..3)
    const int wj     = wid >> 2;       // N wave coord (0..1)
    const int laneN  = lane & 15;
    const int laneHi = lane >> 4;      // 0 / 1

    v8f acc[2][4];
#pragma unroll
    for (int a = 0; a < 2; ++a)
#pragma unroll
        for (int b = 0; b < 4; ++b) acc[a][b] = v8f{};

    // staged-load coordinates: 2 threads per row, 16 K each
    const int lrow = tid >> 1;
    const int lk0  = (tid & 1) * 16;
    const float*  asrc = W_emb + ((size_t)tokS[lrow] * D_MODEL + lk0);
    const __bf16* bsrc = W16T + ((size_t)(ntile * 128 + lrow) * D_MODEL + lk0);

    float4 ar0, ar1, ar2, ar3;
    auto loadA = [&](int kt) {
        const float4* s = reinterpret_cast<const float4*>(asrc + kt * 32);
        ar0 = s[0]; ar1 = s[1]; ar2 = s[2]; ar3 = s[3];
    };
    auto storeA = [&](int buf) {
        __bf16* d = &Al[buf][lrow * 40 + lk0];
        bf16x8 lo, hi;
        lo[0]=(__bf16)ar0.x; lo[1]=(__bf16)ar0.y; lo[2]=(__bf16)ar0.z; lo[3]=(__bf16)ar0.w;
        lo[4]=(__bf16)ar1.x; lo[5]=(__bf16)ar1.y; lo[6]=(__bf16)ar1.z; lo[7]=(__bf16)ar1.w;
        hi[0]=(__bf16)ar2.x; hi[1]=(__bf16)ar2.y; hi[2]=(__bf16)ar2.z; hi[3]=(__bf16)ar2.w;
        hi[4]=(__bf16)ar3.x; hi[5]=(__bf16)ar3.y; hi[6]=(__bf16)ar3.z; hi[7]=(__bf16)ar3.w;
        *reinterpret_cast<bf16x8*>(d)     = lo;
        *reinterpret_cast<bf16x8*>(d + 8) = hi;
    };
    auto issueB = [&](int kt, int buf) {
        const __bf16* g = bsrc + kt * 32;
        __bf16* d = &Bl[buf][lrow * 40 + lk0];
#if USE_ASYNC_LDS
        async_b128(g, d);
        async_b128(g + 8, d + 8);
#else
        const bf16x8* s = reinterpret_cast<const bf16x8*>(g);
        *reinterpret_cast<bf16x8*>(d)     = s[0];
        *reinterpret_cast<bf16x8*>(d + 8) = s[1];
#endif
    };

    // ---- pipeline prologue ----
    loadA(0);
    issueB(0, 0);
    storeA(0);
    async_wait0();
    __syncthreads();

    for (int kt = 0; kt < 32; ++kt) {
        const int cur = kt & 1;
        if (kt < 31) {
            loadA(kt + 1);           // global gather for next step issues early
            issueB(kt + 1, cur ^ 1); // async DMA weights into the other buffer
        }
        // ---- fragments + WMMA on current buffer ----
        v16bf bf[4];
#pragma unroll
        for (int nt = 0; nt < 4; ++nt) {
            const __bf16* bp = &Bl[cur][(wj * 64 + nt * 16 + laneN) * 40 + laneHi * 8];
            bf16x8 b0 = *reinterpret_cast<const bf16x8*>(bp);
            bf16x8 b1 = *reinterpret_cast<const bf16x8*>(bp + 16);
#pragma unroll
            for (int i = 0; i < 8; ++i) { bf[nt][i] = b0[i]; bf[nt][i + 8] = b1[i]; }
        }
#pragma unroll
        for (int mt = 0; mt < 2; ++mt) {
            const __bf16* ap = &Al[cur][(wi * 32 + mt * 16 + laneN) * 40 + laneHi * 8];
            bf16x8 a0 = *reinterpret_cast<const bf16x8*>(ap);
            bf16x8 a1 = *reinterpret_cast<const bf16x8*>(ap + 16);
            v16bf af;
#pragma unroll
            for (int i = 0; i < 8; ++i) { af[i] = a0[i]; af[i + 8] = a1[i]; }
#pragma unroll
            for (int nt = 0; nt < 4; ++nt) {
                acc[mt][nt] = __builtin_amdgcn_wmma_f32_16x16x32_bf16(
                    false, af, false, bf[nt], (short)0, acc[mt][nt], false, false);
            }
        }
        if (kt < 31) storeA(cur ^ 1);   // write next A tile (other buffer)
        async_wait0();
        __syncthreads();
    }

    // ---- epilogue: gelu + row-mask, store and/or pooled column sums ----
    float cAcc[4] = {0.f, 0.f, 0.f, 0.f};
#pragma unroll
    for (int mt = 0; mt < 2; ++mt) {
#pragma unroll
        for (int r = 0; r < 8; ++r) {
            const int m = wi * 32 + mt * 16 + laneHi * 8 + r;
            const float msk = (tokS[m] != 0) ? 1.0f : 0.0f;
#pragma unroll
            for (int nt = 0; nt < 4; ++nt) {
                float g = gelu_tanh(acc[mt][nt][r]) * msk;
                if (MODE == 0 || MODE == 2) {
                    const int colg = ntile * 128 + wj * 64 + nt * 16 + laneN;
                    long rowOut;
                    if (MODE == 0) {
                        int rowg = mtile * 128 + m;
                        int n = rowg >> 8;        // Ts = 256
                        int t = rowg & 255;
                        rowOut = (long)n * FULL_T + TK + t;
                    } else {
                        rowOut = (long)mtile * FULL_T + m;
                    }
                    __builtin_nontemporal_store(g, &out[rowOut * D_MODEL + colg]);
                }
                cAcc[nt] += g;
            }
        }
    }

    if (MODE == 0 || MODE == 1) {
#pragma unroll
        for (int nt = 0; nt < 4; ++nt)
            atomicAdd(&colsum[wj * 64 + nt * 16 + laneN], cAcc[nt]);
        __syncthreads();
        if (tid < 128) {
            if (MODE == 1)
                know_sum[(size_t)mtile * D_MODEL + ntile * 128 + tid] = colsum[tid];
            else
                atomicAdd(&ctx_sum[(size_t)(mtile >> 1) * D_MODEL + ntile * 128 + tid],
                          colsum[tid]);
        }
    }
}

// ------------------------------------------------------------------
// Pooled lengths, scaled dot scores, masked argmax, ck_attn, ids
// ------------------------------------------------------------------
__global__ __launch_bounds__(256) void scores_kernel(const int* __restrict__ src,
                                                     const int* __restrict__ know,
                                                     const unsigned char* __restrict__ ckm,
                                                     const int* __restrict__ cs_ids,
                                                     const int* __restrict__ use_flag,
                                                     const float* __restrict__ ctx_sum,
                                                     const float* __restrict__ know_sum,
                                                     int* __restrict__ ids,
                                                     float* __restrict__ out) {
    const int n = blockIdx.x;
    const int tid = threadIdx.x;
    __shared__ float red[256];
    __shared__ int   klen[32];
    __shared__ float sc[32];
    __shared__ int   clen_s;

    // ctx length
    red[tid] = (src[n * TS + tid] != 0) ? 1.0f : 0.0f;
    __syncthreads();
    for (int s = 128; s > 0; s >>= 1) {
        if (tid < s) red[tid] += red[tid + s];
        __syncthreads();
    }
    if (tid == 0) { int c = (int)(red[0] + 0.5f); clen_s = c > 1 ? c : 1; }
    if (tid < 32) klen[tid] = 0;
    __syncthreads();

    // knowledge lengths
    for (int i = tid; i < K_KNOW * TK; i += 256) {
        if (know[n * K_KNOW * TK + i] != 0) atomicAdd(&klen[i >> 7], 1);
    }
    __syncthreads();

    // dot(know_use, ctx_use) with 1/sqrt(d*len) folded in
    const float* cs = ctx_sum + (size_t)n * D_MODEL;
    for (int k = 0; k < K_KNOW; ++k) {
        const float* ks = know_sum + (size_t)(n * K_KNOW + k) * D_MODEL;
        float p = 0.0f;
        for (int col = tid; col < D_MODEL; col += 256) p += ks[col] * cs[col];
        red[tid] = p;
        __syncthreads();
        for (int s = 128; s > 0; s >>= 1) {
            if (tid < s) red[tid] += red[tid + s];
            __syncthreads();
        }
        if (tid == 0) {
            int kl = klen[k] > 1 ? klen[k] : 1;
            sc[k] = red[0] / ((float)D_MODEL * sqrtf((float)kl * (float)clen_s));
        }
        __syncthreads();
    }

    if (tid == 0) {
        int bi = 0;
        float bv = -3.4e38f;
        for (int k = 0; k < K_KNOW; ++k) {
            float v = ckm[n * K_KNOW + k] ? sc[k] : NEG_INF_F;
            if (v > bv) { bv = v; bi = k; }           // first max, matching jnp.argmax
        }
        ids[n] = (*use_flag) ? cs_ids[n] : bi;
    }
    if (tid < 32) {
        float v = ckm[n * K_KNOW + tid] ? sc[tid] : 0.0f;
        out[OUT_ENC_ELEMS + OUT_MASK_ELEMS + (size_t)n * K_KNOW + tid] = v;
    }
}

__global__ __launch_bounds__(384) void mask_kernel(const int* __restrict__ src,
                                                   const int* __restrict__ know,
                                                   const int* __restrict__ ids,
                                                   float* __restrict__ out) {
    const int n = blockIdx.x;
    const int t = threadIdx.x;                 // 0..383
    float v;
    if (t < TK) v = (know[(n * K_KNOW + ids[n]) * TK + t] != 0) ? 1.0f : 0.0f;
    else        v = (src[n * TS + (t - TK)] != 0) ? 1.0f : 0.0f;
    out[OUT_ENC_ELEMS + (size_t)n * FULL_T + t] = v;
}

extern "C" void kernel_launch(void* const* d_in, const int* in_sizes, int n_in,
                              void* d_out, int out_size, void* d_ws, size_t ws_size,
                              hipStream_t stream) {
    (void)in_sizes; (void)n_in; (void)out_size; (void)ws_size;
    const int*           src_tokens  = (const int*)d_in[0];
    const int*           know_tokens = (const int*)d_in[1];
    const unsigned char* ck_mask     = (const unsigned char*)d_in[2];
    const int*           cs_ids      = (const int*)d_in[3];
    const int*           use_cs_ids  = (const int*)d_in[4];
    const float*         W_emb       = (const float*)d_in[5];
    const float*         W_proj      = (const float*)d_in[6];
    float*               out         = (float*)d_out;

    __bf16* W16T     = (__bf16*)((char*)d_ws + WS_W16T_OFF);
    float*  ctx_sum  = (float*)((char*)d_ws + WS_CTXSUM_OFF);
    float*  know_sum = (float*)((char*)d_ws + WS_KNOWSUM_OFF);
    int*    ids      = (int*)((char*)d_ws + WS_IDS_OFF);

    // 1. one-time bf16 transpose of W_proj
    convert_w_kernel<<<4096, 256, 0, stream>>>(W_proj, W16T);
    // 2. zero ctx pooled-sum accumulator
    zero_ctx_kernel<<<32, 256, 0, stream>>>(ctx_sum);
    // 3. context encode: 2048 rows -> 16 M-tiles x 8 N-tiles
    encode_gemm<0><<<dim3(16, 8), 256, 0, stream>>>(src_tokens, W_emb, W16T, out,
                                                    ctx_sum, nullptr, nullptr);
    // 4. knowledge encode + in-block pooling: 256 chunks x 8 N-tiles
    encode_gemm<1><<<dim3(256, 8), 256, 0, stream>>>(know_tokens, W_emb, W16T, nullptr,
                                                     nullptr, know_sum, nullptr);
    // 5. scores / argmax / ck_attn / ids
    scores_kernel<<<8, 256, 0, stream>>>(src_tokens, know_tokens, ck_mask, cs_ids,
                                         use_cs_ids, ctx_sum, know_sum, ids, out);
    // 6. re-encode only the selected chunk per batch item
    encode_gemm<2><<<dim3(8, 8), 256, 0, stream>>>(know_tokens, W_emb, W16T, out,
                                                   nullptr, nullptr, ids);
    // 7. full_mask
    mask_kernel<<<8, 384, 0, stream>>>(src_tokens, know_tokens, ids, out);
}
